// N3Aggregation2D_71511205478648
// MI455X (gfx1250) — matrix operand order repaired and checked by reading, not compiled, not run on hardware
//
#include <hip/hip_runtime.h>

// ---------------------------------------------------------------------------
// N3Net forward for MI455X (gfx1250, wave32, WMMA).
//
//  K0  pack ye/xe to f16 pixel-major [t][y][x][e] (32B/pixel), x -> xp144
//  K1  per-pixel squared norms + separable 10x10 boxsums -> Sy, Sx
//  K3  WMMA patch-GEMM: C[u,v] = <Y_patch(u), X_patch(v)>, K=1600,
//      v_wmma_f32_16x16x32_f16 with K-chunk k = e + 16*px_off so both A and B
//      fragments are contiguous LDS b128 loads (no VALU repacking).
//      Fused epilogue: dist = Sy + Sx - 2C, self -> -BIG, negd[O][t][121][121]
//  K4  per-query softmax stats (max + 7 iterative-exclusion denominators)
//  K5  per dy-group (15 offsets): regenerate W fields, separable fold,
//      register-resident z accumulation (112 acc / thread)
//  K6  normalize by closed-form coverage, subtract y0, concat, crop
//
// Scratch requirement: ~148 MB in d_ws (all intermediates L2-resident).
// ---------------------------------------------------------------------------

typedef _Float16 f16;
typedef __attribute__((ext_vector_type(16))) _Float16     v16h;
typedef __attribute__((ext_vector_type(8)))  float        v8f;
typedef __attribute__((ext_vector_type(4)))  unsigned int v4u;

#define T_    4
#define E_    16
#define C_    16
#define HI    128      // original h/w
#define HP    130      // padded by 1
#define HE    144      // padded by 1 + r(7) each side
#define NQ    121      // nH = nW = HP - PS + 1
#define NV    135      // database extent NQ + 2r
#define PS    10
#define R_    7
#define WS    15
#define O_    225
#define KNN   7
#define BIGD  1e30f
#define OSELF 112      // (0,0) offset index: (7)*15+7

// ---------------------------------------------------------------- pack / norms

// dst pixel-major: [T][dim][dim][E_]
__global__ void k_pack_half(const float* __restrict__ src, f16* __restrict__ dst,
                            int dim, int pad) {
  long i = (long)blockIdx.x * blockDim.x + threadIdx.x;
  long tot = (long)T_ * E_ * dim * dim;
  if (i >= tot) return;
  int e = i % E_; long r = i / E_;
  int ix = r % dim; r /= dim;
  int iy = r % dim; int t = (int)(r / dim);
  int oy = iy - pad, ox = ix - pad;
  float v = 0.f;
  if (oy >= 0 && oy < HI && ox >= 0 && ox < HI)
    v = src[((long)(t * E_ + e) * HI + oy) * HI + ox];
  dst[i] = (f16)v;
}

__global__ void k_pack_float(const float* __restrict__ src, float* __restrict__ dst,
                             int dim, int pad) {
  long i = (long)blockIdx.x * blockDim.x + threadIdx.x;
  long tot = (long)T_ * C_ * dim * dim;
  if (i >= tot) return;
  int ix = i % dim; long r = i / dim;
  int iy = r % dim; r /= dim;
  int ch = r % C_;  int t = (int)(r / C_);
  int oy = iy - pad, ox = ix - pad;
  float v = 0.f;
  if (oy >= 0 && oy < HI && ox >= 0 && ox < HI)
    v = src[((long)(t * C_ + ch) * HI + oy) * HI + ox];
  dst[i] = v;
}

__global__ void k_pix_norm(const float* __restrict__ src, float* __restrict__ dst,
                           int dim, int pad) {
  long i = (long)blockIdx.x * blockDim.x + threadIdx.x;
  long tot = (long)T_ * dim * dim;
  if (i >= tot) return;
  int ix = i % dim; long r = i / dim;
  int iy = r % dim; int t = (int)(r / dim);
  int oy = iy - pad, ox = ix - pad;
  float s = 0.f;
  if (oy >= 0 && oy < HI && ox >= 0 && ox < HI) {
    const float* p = src + ((long)t * E_ * HI + oy) * HI + ox;
    for (int e = 0; e < E_; ++e) { float v = p[(long)e * HI * HI]; s += v * v; }
  }
  dst[i] = s;
}

// valid 10-tap row sum: in[t][H][Win] -> out[t][H][Win-9]
__global__ void k_row_box(const float* __restrict__ in, float* __restrict__ out,
                          int H, int Win) {
  int Wout = Win - PS + 1;
  long i = (long)blockIdx.x * blockDim.x + threadIdx.x;
  long tot = (long)T_ * H * Wout;
  if (i >= tot) return;
  int x = i % Wout; long r = i / Wout;
  int y = r % H; int t = (int)(r / H);
  const float* p = in + ((long)t * H + y) * Win + x;
  float s = 0.f;
  for (int px = 0; px < PS; ++px) s += p[px];
  out[i] = s;
}

// valid 10-tap col sum: in[t][Hin][W] -> out[t][Hin-9][W]
__global__ void k_col_box(const float* __restrict__ in, float* __restrict__ out,
                          int Hin, int W) {
  int Hout = Hin - PS + 1;
  long i = (long)blockIdx.x * blockDim.x + threadIdx.x;
  long tot = (long)T_ * Hout * W;
  if (i >= tot) return;
  int x = i % W; long r = i / W;
  int y = r % Hout; int t = (int)(r / Hout);
  const float* p = in + ((long)t * Hin + y) * W + x;
  float s = 0.f;
  for (int py = 0; py < PS; ++py) s += p[(long)py * W];
  out[i] = s;
}

// --------------------------------------------------------- K3: WMMA patch GEMM
// Block = 128 threads = 4 wave32; wave w handles query row uy = uy0 + w,
// 16 queries ux0..ux0+15. Per vy in [uy-7,uy+7], two band halves
// (vx0 = ux0-7 + 16b) accumulate simultaneously, sharing A fragments.
//
// K-chunk (32) for fixed (py, pxb): k = e + 16*px_off, px in {pxb, pxb+1}.
// With pixel-major LDS tiles ([row][col][16 channels], 32B/pixel):
//   A frag lane (grp,m): h[0..7]  = tile(m+pxb+grp)[0..7]   (b128)
//                        h[8..15] = tile(m+pxb+grp)[8..15]   (b128)
//   B frag lane (grp,n): h[0..15] = tile(n+pxb+grp)[0..15]            (2x b128)

__global__ __launch_bounds__(128) void k_search_wmma(
    const f16* __restrict__ yeh,    // [T][130][130][16]
    const f16* __restrict__ xeh,    // [T][144][144][16]
    const float* __restrict__ Sy,   // [T][121][121]
    const float* __restrict__ Sx,   // [T][135][135]
    float* __restrict__ negd)       // [225][T][121][121]
{
  __shared__ __align__(16) f16 sYe[13][26][E_];  // rows uy0..+12, cols ux0..+25
  __shared__ __align__(16) f16 sXe[27][42][E_];  // rows uy0-7.., cols ux0-7..

  const int t   = blockIdx.z;
  const int uy0 = blockIdx.y * 4;
  const int ux0 = blockIdx.x * 16;

  // cooperative staging: one 32B pixel per iteration (b128 in, b128 out)
  for (int pix = threadIdx.x; pix < 13 * 26; pix += 128) {
    int rx = pix % 26, ry = pix / 26;
    int gy = uy0 + ry, gx = ux0 + rx;
    v4u a0 = {0u, 0u, 0u, 0u}, a1 = {0u, 0u, 0u, 0u};
    if (gy < HP && gx < HP) {
      const v4u* gp = (const v4u*)(yeh + (((long)t * HP + gy) * HP + gx) * E_);
      a0 = gp[0]; a1 = gp[1];
    }
    v4u* sp = (v4u*)(&sYe[ry][rx][0]);
    sp[0] = a0; sp[1] = a1;
  }
  for (int pix = threadIdx.x; pix < 27 * 42; pix += 128) {
    int rx = pix % 42, ry = pix / 42;
    int gy = uy0 + ry, gx = ux0 + rx;   // 144-coord shift cancels the -7
    v4u a0 = {0u, 0u, 0u, 0u}, a1 = {0u, 0u, 0u, 0u};
    if (gy < HE && gx < HE) {
      const v4u* gp = (const v4u*)(xeh + (((long)t * HE + gy) * HE + gx) * E_);
      a0 = gp[0]; a1 = gp[1];
    }
    v4u* sp = (v4u*)(&sXe[ry][rx][0]);
    sp[0] = a0; sp[1] = a1;
  }
  __syncthreads();

  const int w  = threadIdx.x >> 5;
  const int uy = uy0 + w;
  if (uy >= NQ) return;                   // wave-uniform: EXEC stays all-ones
  const int lane = threadIdx.x & 31;
  const int grp  = lane >> 4;
  const int mn   = lane & 15;             // A: row m ; B/C: col n
  const int eOff = 8 * grp;

  for (int vyi = 0; vyi < WS; ++vyi) {
    const int vy = uy - R_ + vyi;
    v8f acc0 = {0.f, 0.f, 0.f, 0.f, 0.f, 0.f, 0.f, 0.f};
    v8f acc1 = {0.f, 0.f, 0.f, 0.f, 0.f, 0.f, 0.f, 0.f};
    for (int py = 0; py < PS; ++py) {
      const int ryA = (uy - uy0) + py;
      const int ryB = vy + py - uy0 + 7;
      for (int pxb = 0; pxb < PS; pxb += 2) {
        union { v16h v; v4u q[2]; } Af, B0, B1;
        const int colA = mn + pxb;
        Af.q[0] = *(const v4u*)(&sYe[ryA][colA + grp][0]);
        Af.q[1] = *(const v4u*)(&sYe[ryA][colA + grp][8]);
        const int colB = mn + pxb + grp;
        const v4u* bp0 = (const v4u*)(&sXe[ryB][colB][0]);
        B0.q[0] = bp0[0]; B0.q[1] = bp0[1];
        const v4u* bp1 = (const v4u*)(&sXe[ryB][colB + 16][0]);
        B1.q[0] = bp1[0]; B1.q[1] = bp1[1];
        acc0 = __builtin_amdgcn_wmma_f32_16x16x32_f16(
            false, Af.v, false, B0.v, (short)0, acc0, false, false);
        acc1 = __builtin_amdgcn_wmma_f32_16x16x32_f16(
            false, Af.v, false, B1.v, (short)0, acc1, false, false);
      }
    }
    // scatter: acc VGPR i, lane -> (m = i + 8*grp, n = mn)
#pragma unroll
    for (int b = 0; b < 2; ++b) {
#pragma unroll
      for (int i = 0; i < 8; ++i) {
        const int m  = i + 8 * grp;
        const int ux = ux0 + m;
        if (ux >= NQ) continue;
        const int dx = mn - m + (b ? 9 : -7);
        if (dx < -R_ || dx > R_) continue;
        const int o  = (vy - uy + R_) * WS + (dx + R_);
        const float cv = b ? acc1[i] : acc0[i];
        float d;
        if (o == OSELF) {
          d = BIGD;                       // remove_self
        } else {
          const int vx = ux + dx;
          float sy = Sy[((long)t * NQ + uy) * NQ + ux];
          float sx = Sx[((long)t * NV + (vy + R_)) * NV + (vx + R_)];
          d = sy + sx - 2.0f * cv;
        }
        negd[(((long)o * T_ + t) * NQ + uy) * NQ + ux] = -d;
      }
    }
  }
}

// --------------------------------------------------- K4: per-query softmax stats
// stats[q] = { M = max_j l_j,  S_1..S_7 } with a^{i+1} = a^i * max(1 - a^i/S_i, 1e-8)

__global__ void k_stats(const float* __restrict__ negd,
                        const float* __restrict__ logt,   // [T][1][128][128]
                        float* __restrict__ stats) {
  long i = (long)blockIdx.x * blockDim.x + threadIdx.x;
  long tot = (long)T_ * NQ * NQ;
  if (i >= tot) return;
  int ux = i % NQ; long r = i / NQ;
  int uy = r % NQ; int t = (int)(r / NQ);
  float lt   = logt[((long)t * HI + (uy + PS / 2 - 1)) * HI + (ux + PS / 2 - 1)];
  float invT = __expf(-lt);
  const long qs = (long)T_ * NQ * NQ;
  const float* nd = negd + (long)t * NQ * NQ + (long)uy * NQ + ux;

  float M = -3.4e38f;
  for (int o = 0; o < O_; ++o) M = fmaxf(M, nd[(long)o * qs] * invT);

  float S[KNN];
  for (int it = 0; it < KNN; ++it) {
    float s = 0.f;
    for (int o = 0; o < O_; ++o) {
      float a = __expf(nd[(long)o * qs] * invT - M);
      for (int p = 0; p < it; ++p) a *= fmaxf(1.f - a / S[p], 1e-8f);
      s += a;
    }
    S[it] = s;
  }
  float* st = stats + i * 8;
  st[0] = M;
  for (int it = 0; it < KNN; ++it) st[1 + it] = S[it];
}

// ------------------------------------------- K5a: weight fields for dy-group g

__global__ void k_wfield(const float* __restrict__ negd,
                         const float* __restrict__ logt,
                         const float* __restrict__ stats,
                         float* __restrict__ Wf, int g) {
  long i = (long)blockIdx.x * blockDim.x + threadIdx.x;
  long tot = (long)WS * T_ * NQ * NQ;
  if (i >= tot) return;
  int ux = i % NQ; long r = i / NQ;
  int uy = r % NQ; r /= NQ;
  int t = r % T_;  int s = (int)(r / T_);
  const int o = g * WS + s;
  float lt   = logt[((long)t * HI + (uy + PS / 2 - 1)) * HI + (ux + PS / 2 - 1)];
  float invT = __expf(-lt);
  const float* st = stats + ((long)t * NQ * NQ + (long)uy * NQ + ux) * 8;
  float a = __expf(negd[(((long)o * T_ + t) * NQ + uy) * NQ + ux] * invT - st[0]);
  const long qb = (long)uy * NQ + ux;
  for (int it = 0; it < KNN; ++it) {
    float Wv = a / st[1 + it];
    Wf[(((long)s * T_ + t) * KNN + it) * ((long)NQ * NQ) + qb] = Wv;
    a *= fmaxf(1.f - Wv, 1e-8f);
  }
}

// -------------------------- K5b: row fold (full boxsum, rows): Wf -> Rb [..][121][130]

__global__ void k_rowfold(const float* __restrict__ Wf, float* __restrict__ Rb) {
  long i = (long)blockIdx.x * blockDim.x + threadIdx.x;
  long tot = (long)WS * T_ * KNN * NQ * HP;
  if (i >= tot) return;
  int x = i % HP; long r = i / HP;
  int uy = r % NQ; r /= NQ;
  int it = r % KNN; r /= KNN;
  int t = r % T_;  int s = (int)(r / T_);
  const float* wp = Wf + (((long)s * T_ + t) * KNN + it) * ((long)NQ * NQ) + (long)uy * NQ;
  int lo = max(x - (PS - 1), 0), hi = min(x, NQ - 1);
  float sum = 0.f;
  for (int u = lo; u <= hi; ++u) sum += wp[u];
  Rb[i] = sum;
}

// ------------- K5c: col fold + z accumulation (register-resident 7x16 per pixel)

__global__ __launch_bounds__(128) void k_accum(const float* __restrict__ Rb,
                                               const float* __restrict__ xp,   // [T][C][144][144]
                                               float* __restrict__ z,          // [T][7][16][130][130]
                                               int g) {
  long i = (long)blockIdx.x * blockDim.x + threadIdx.x;
  long tot = (long)T_ * HP * HP;
  if (i >= tot) return;
  int x = i % HP; long r = i / HP;
  int y = r % HP; int t = (int)(r / HP);
  const int dy = g - R_;

  float zr[KNN][C_];
  const long zbase = (long)t * KNN * C_ * HP * HP + (long)y * HP + x;
  if (g == 0) {
    for (int it = 0; it < KNN; ++it)
      for (int c = 0; c < C_; ++c) zr[it][c] = 0.f;
  } else {
    for (int it = 0; it < KNN; ++it)
      for (int c = 0; c < C_; ++c)
        zr[it][c] = z[zbase + (long)(it * C_ + c) * HP * HP];
  }

  const int ulo = max(y - (PS - 1), 0), uhi = min(y, NQ - 1);
  for (int s = 0; s < WS; ++s) {
    const int dx = s - R_;
    if (s + 1 < WS)  // hint next offset's fold rows into cache (global_prefetch_b8)
      __builtin_prefetch(Rb + ((((long)(s + 1) * T_ + t) * KNN) * NQ + ulo) * HP + x, 0, 1);
    float xv[C_];
    for (int c = 0; c < C_; ++c)
      xv[c] = xp[(((long)t * C_ + c) * HE + (y + dy + R_)) * HE + (x + dx + R_)];
    float bw[KNN];
    for (int it = 0; it < KNN; ++it) {
      const float* rb = Rb + ((((long)s * T_ + t) * KNN + it) * NQ) * HP + x;
      float ssum = 0.f;
      for (int u = ulo; u <= uhi; ++u) ssum += rb[(long)u * HP];
      bw[it] = ssum;
    }
    for (int it = 0; it < KNN; ++it)
      for (int c = 0; c < C_; ++c)
        zr[it][c] = fmaf(bw[it], xv[c], zr[it][c]);
  }

  for (int it = 0; it < KNN; ++it)
    for (int c = 0; c < C_; ++c)
      z[zbase + (long)(it * C_ + c) * HP * HP] = zr[it][c];
}

// ------------------------------------------------------------- K6: epilogue

__device__ inline int covcnt(int v) { return min(v, NQ - 1) - max(v - (PS - 1), 0) + 1; }

__global__ void k_final(const float* __restrict__ x_in,
                        const float* __restrict__ z,
                        float* __restrict__ out) {
  long i = (long)blockIdx.x * blockDim.x + threadIdx.x;
  long tot = (long)T_ * (C_ * (KNN + 1)) * HI * HI;
  if (i >= tot) return;
  int ox = i % HI; long r = i / HI;
  int oy = r % HI; r /= HI;
  int ch = r % (C_ * (KNN + 1));
  int t = (int)(r / (C_ * (KNN + 1)));
  float v;
  if (ch < C_) {
    v = x_in[((long)(t * C_ + ch) * HI + oy) * HI + ox];   // y0 cropped == x
  } else {
    int kk = (ch - C_) / C_, c = (ch - C_) % C_;
    int y = oy + 1, x = ox + 1;
    float wn = (float)(covcnt(y) * covcnt(x));
    float zz = z[(((long)t * KNN + kk) * C_ + c) * ((long)HP * HP) + (long)y * HP + x];
    v = zz / (wn + 1e-10f) - x_in[((long)(t * C_ + c) * HI + oy) * HI + ox];
  }
  out[i] = v;
}

// ---------------------------------------------------------------------------

extern "C" void kernel_launch(void* const* d_in, const int* in_sizes, int n_in,
                              void* d_out, int out_size, void* d_ws, size_t ws_size,
                              hipStream_t stream) {
  (void)in_sizes; (void)n_in; (void)out_size; (void)ws_size;
  const float* x  = (const float*)d_in[0];
  const float* xe = (const float*)d_in[1];
  const float* ye = (const float*)d_in[2];
  const float* lt = (const float*)d_in[3];
  float* out = (float*)d_out;

  char* p = (char*)d_ws;
  auto carve = [&](size_t bytes) -> char* {
    char* q = p; p += (bytes + 255) & ~size_t(255); return q;
  };
  f16*   yeh   = (f16*)  carve((size_t)T_ * E_ * HP * HP * 2);
  f16*   xeh   = (f16*)  carve((size_t)T_ * E_ * HE * HE * 2);
  float* xp    = (float*)carve((size_t)T_ * C_ * HE * HE * 4);
  float* ny    = (float*)carve((size_t)T_ * HP * HP * 4);
  float* nx    = (float*)carve((size_t)T_ * HE * HE * 4);
  float* rsy   = (float*)carve((size_t)T_ * HP * NQ * 4);
  float* rsx   = (float*)carve((size_t)T_ * HE * NV * 4);
  float* Sy    = (float*)carve((size_t)T_ * NQ * NQ * 4);
  float* Sx    = (float*)carve((size_t)T_ * NV * NV * 4);
  float* negd  = (float*)carve((size_t)O_ * T_ * NQ * NQ * 4);
  float* stats = (float*)carve((size_t)T_ * NQ * NQ * 8 * 4);
  float* Wf    = (float*)carve((size_t)WS * T_ * KNN * NQ * NQ * 4);
  float* Rb    = (float*)carve((size_t)WS * T_ * KNN * NQ * HP * 4);
  float* zacc  = (float*)carve((size_t)T_ * KNN * C_ * HP * HP * 4);

  const int B = 256;
  auto g1 = [&](long n) { return dim3((unsigned)((n + B - 1) / B)); };

  k_pack_half <<<g1((long)T_ * E_ * HP * HP), B, 0, stream>>>(ye, yeh, HP, 1);
  k_pack_half <<<g1((long)T_ * E_ * HE * HE), B, 0, stream>>>(xe, xeh, HE, 8);
  k_pack_float<<<g1((long)T_ * C_ * HE * HE), B, 0, stream>>>(x,  xp,  HE, 8);
  k_pix_norm  <<<g1((long)T_ * HP * HP), B, 0, stream>>>(ye, ny, HP, 1);
  k_pix_norm  <<<g1((long)T_ * HE * HE), B, 0, stream>>>(xe, nx, HE, 8);
  k_row_box   <<<g1((long)T_ * HP * NQ), B, 0, stream>>>(ny, rsy, HP, HP);
  k_col_box   <<<g1((long)T_ * NQ * NQ), B, 0, stream>>>(rsy, Sy, HP, NQ);
  k_row_box   <<<g1((long)T_ * HE * NV), B, 0, stream>>>(nx, rsx, HE, HE);
  k_col_box   <<<g1((long)T_ * NV * NV), B, 0, stream>>>(rsx, Sx, HE, NV);

  dim3 gs((NQ + 15) / 16, (NQ + 3) / 4, T_);       // 8 x 31 x 4
  k_search_wmma<<<gs, 128, 0, stream>>>(yeh, xeh, Sy, Sx, negd);

  k_stats<<<g1((long)T_ * NQ * NQ), B, 0, stream>>>(negd, lt, stats);

  for (int g = 0; g < WS; ++g) {
    k_wfield <<<g1((long)WS * T_ * NQ * NQ), B, 0, stream>>>(negd, lt, stats, Wf, g);
    k_rowfold<<<g1((long)WS * T_ * KNN * NQ * HP), B, 0, stream>>>(Wf, Rb);
    k_accum  <<<g1((long)T_ * HP * HP), 128, 0, stream>>>(Rb, xp, zacc, g);
  }

  k_final<<<g1((long)T_ * C_ * (KNN + 1) * HI * HI), B, 0, stream>>>(x, zacc, out);
}
